// MoEResidualAttentionBlock_70377334113064
// MI455X (gfx1250) — matrix operand order repaired and checked by reading, not compiled, not run on hardware
//
#include <hip/hip_runtime.h>
#include <cstdint>
#include <cstddef>

// ---------------------------------------------------------------------------
// MoE Residual Attention Block for gfx1250 (MI455X), bf16 WMMA everywhere.
// S=1024 B=8 D=768 H=12 hd=64 E=8 K(top)=2 FF=3072, T = S*B = 8192 tokens.
// ---------------------------------------------------------------------------

#define DEV static __device__ __forceinline__

typedef __attribute__((ext_vector_type(16))) __bf16 bf16x16;
typedef __attribute__((ext_vector_type(8)))  float  f32x8;

union AFrag { bf16x16 v; __bf16 e[16]; };
union CFrag { f32x8   v; float  e[8];  };

static constexpr int Sdim = 1024, Bdim = 8, Ddim = 768, Hn = 12, En = 8, FFdim = 3072;
static constexpr int Tn = Sdim * Bdim;   // 8192 tokens
static constexpr int HD = 64;            // head dim

DEV __bf16 f2bf(float f) { return (__bf16)f; }
DEV float  bf2f(__bf16 h) { return (float)h; }
DEV int lane_id() { return threadIdx.x & 31; }

DEV f32x8 wmma_bf16(bf16x16 a, bf16x16 b, f32x8 c) {
  return __builtin_amdgcn_wmma_f32_16x16x32_bf16(false, a, false, b, (short)0, c,
                                                 false, false);
}

// ---- VALU cross-lane reductions via v_permlane16_b32 / v_permlanex16_b32 ----
// (avoids ds_bpermute + DScnt waits that __shfl_xor generates on wave32)
DEV float pl16(float v, int s0, int s1) {
  int x = __float_as_int(v);
  return __int_as_float(__builtin_amdgcn_permlane16(x, x, s0, s1, false, false));
}
DEV float plx16(float v) {   // lane i <-> lane i^16
  int x = __float_as_int(v);
  return __int_as_float(
      __builtin_amdgcn_permlanex16(x, x, 0x76543210, 0xFEDCBA98, false, false));
}
// xor-mask nibble-select tables within a 16-lane row
#define SEL_X1_LO 0x67452301
#define SEL_X1_HI 0xEFCDAB89
#define SEL_X2_LO 0x54761032
#define SEL_X2_HI 0xDCFE98BA
#define SEL_X4_LO 0x32107654
#define SEL_X4_HI 0xBA98FEDC
#define SEL_X8_LO 0xFEDCBA98
#define SEL_X8_HI 0x76543210

DEV float rmax16(float v) {
  v = fmaxf(v, pl16(v, SEL_X1_LO, SEL_X1_HI));
  v = fmaxf(v, pl16(v, SEL_X2_LO, SEL_X2_HI));
  v = fmaxf(v, pl16(v, SEL_X4_LO, SEL_X4_HI));
  v = fmaxf(v, pl16(v, SEL_X8_LO, SEL_X8_HI));
  return v;
}
DEV float rsum16(float v) {
  v += pl16(v, SEL_X1_LO, SEL_X1_HI);
  v += pl16(v, SEL_X2_LO, SEL_X2_HI);
  v += pl16(v, SEL_X4_LO, SEL_X4_HI);
  v += pl16(v, SEL_X8_LO, SEL_X8_HI);
  return v;
}
DEV float rsum32(float v) {
  v = rsum16(v);
  v += plx16(v);
  return v;
}

// A-frag 16x32 (MxK) bf16, row-major source, element stride `stride`.
DEV AFrag load_a_frag(const __bf16* src, int stride) {
  AFrag f; int l = lane_id(); int m = l & 15; int hi = (l >> 4) & 1;
  const __bf16* p = src + m * stride + hi * 8;
#pragma unroll
  for (int i = 0; i < 8; ++i) { f.e[i] = p[i]; f.e[i + 8] = p[i + 16]; }
  return f;
}
// B-frag 32x16 (KxN): element (k=hi*8+i+(i&8), n=lane&15) at src[k*stride + n]
DEV bf16x16 load_b_frag(const __bf16* src, int stride) {
  AFrag f; int l = lane_id(); int n = l & 15; int hi = (l >> 4) & 1;
#pragma unroll
  for (int i = 0; i < 16; ++i) f.e[i] = src[(hi * 8 + i + (i & 8)) * stride + n];
  return f.v;
}
// B-frag from transposed storage: element (k,n) at src[n*stride + k]
DEV bf16x16 load_bT_frag(const __bf16* src, int stride) {
  AFrag f; int l = lane_id(); int n = l & 15; int hi = (l >> 4) & 1;
  const __bf16* p = src + n * stride + hi * 8;
#pragma unroll
  for (int i = 0; i < 16; ++i) f.e[i] = p[i + (i & 8)];
  return f.v;
}

// ---------------------------------------------------------------------------
// fp32 -> bf16 conversion
// ---------------------------------------------------------------------------
__global__ void f32_to_bf16_kernel(const float* __restrict__ in,
                                   __bf16* __restrict__ out, long n) {
  long i = (long)blockIdx.x * blockDim.x + threadIdx.x;
  long stride = (long)gridDim.x * blockDim.x;
  for (; i < n; i += stride) out[i] = f2bf(in[i]);
}

// ---------------------------------------------------------------------------
// LayerNorm: one wave per row of 768 (24 elems / lane), fp32 in, bf16 out
// ---------------------------------------------------------------------------
__launch_bounds__(256)
__global__ void ln_kernel(const float* __restrict__ x, const float* __restrict__ w,
                          const float* __restrict__ b, __bf16* __restrict__ out) {
  int wv = threadIdx.x >> 5, lane = threadIdx.x & 31;
  int row = blockIdx.x * 8 + wv;
  const float* p = x + (size_t)row * Ddim;
  float v[24]; float s = 0.f;
#pragma unroll
  for (int j = 0; j < 24; ++j) { v[j] = p[lane + 32 * j]; s += v[j]; }
  s = rsum32(s);
  float mean = s * (1.0f / Ddim);
  float q = 0.f;
#pragma unroll
  for (int j = 0; j < 24; ++j) { float d = v[j] - mean; q += d * d; }
  q = rsum32(q);
  float inv = rsqrtf(q * (1.0f / Ddim) + 1e-5f);
  __bf16* o = out + (size_t)row * Ddim;
#pragma unroll
  for (int j = 0; j < 24; ++j) {
    int d = lane + 32 * j;
    o[d] = f2bf((v[j] - mean) * inv * w[d] + b[d]);
  }
}

// ---------------------------------------------------------------------------
// Generic 64x128-tile bf16 WMMA GEMM, 256 threads (8 waves), double-buffered.
//  MODE 0: qkv    C[T,2304] = h * Wqkv^T  (+bias, scale q, scatter to q/k/v)
//  MODE 1: outprj C[T,768]  = o * Wout^T  (+bias + residual -> x1 fp32)
//  MODE 2: ffn1   gathered rows of h2 * w1[e]  (+b1, QuickGELU -> mid bf16)
//  MODE 3: ffn2   mid rows * w2[e]  (* combine weight + b2 -> slots fp32)
// ---------------------------------------------------------------------------
struct GArgs {
  const __bf16* A;
  const __bf16* B;          // modes 0/1: [N,K] (transposed); modes 2/3: [E,K,N]
  const float* bias;        // modes 2/3: per-expert slices [E,N]
  const float* resid;       // mode 1
  __bf16* qo; __bf16* ko; __bf16* vo;   // mode 0
  float*  out_f32;          // mode 1: x1 ; mode 3: slots
  __bf16* out_bf16;         // mode 2: mid
  const int* rows; const float* wts; const int* cnt; const int* off;
  int N, K, lda, ldb;
};

static constexpr int AS = 34;    // LDS stride (halves): 17 words (odd, conflict-free)
static constexpr int BS = 130;   // LDS stride: 65 words (odd)

template <int MODE>
__launch_bounds__(256)
__global__ void gemm_kernel(GArgs g) {
  __shared__ __bf16 As[2][64 * AS];
  __shared__ __bf16 Bs[2][32 * BS];
  const int tid = threadIdx.x;
  const int wv = tid >> 5;
  const int lane = tid & 31;
  const int mb = blockIdx.y * 64;
  const int nb = blockIdx.x * 128;

  int base = 0, cnt_e = 0;
  const __bf16* Bm = g.B;
  const float* bias = g.bias;
  if constexpr (MODE >= 2) {
    int e = blockIdx.z;
    base = g.off[e]; cnt_e = g.cnt[e];
    if (mb >= cnt_e) return;                 // uniform early exit
    Bm = g.B + (size_t)e * g.K * g.N;
    bias = g.bias + (size_t)e * g.N;
  }

  // per-thread A staging assignment: 4 threads per row, 8 halves (16B) each
  const int ar = tid >> 2;     // 0..63
  const int aseg = tid & 3;    // 0..3
  const __bf16* Arow;
  if constexpr (MODE <= 1) {
    Arow = g.A + (size_t)(mb + ar) * g.lda;
  } else if constexpr (MODE == 2) {
    int lm = mb + ar;
    int tok = (lm < cnt_e) ? (g.rows[base + lm] >> 1) : 0;
    Arow = g.A + (size_t)tok * g.lda;
  } else {
    int lm = mb + ar;
    int p = (lm < cnt_e) ? (base + lm) : base;
    Arow = g.A + (size_t)p * g.lda;
  }

  auto stage = [&](int buf, int k0) {
    // A (64 x 32)
    {
      const uint4 q = *(const uint4*)(Arow + k0 + aseg * 8);
      unsigned int* dst = (unsigned int*)(&As[buf][0] + ar * AS + aseg * 8);
      dst[0] = q.x; dst[1] = q.y; dst[2] = q.z; dst[3] = q.w;
      if (k0 + 32 < g.K) __builtin_prefetch(Arow + k0 + 32 + aseg * 8, 0, 1);
    }
    // B (32 x 128)
    if constexpr (MODE <= 1) {               // W stored [N,K]: transpose into LDS
#pragma unroll
      for (int it = 0; it < 2; ++it) {
        int id = tid + it * 256;
        int n = id >> 2, seg = id & 3;
        const uint4 q = *(const uint4*)(Bm + (size_t)(nb + n) * g.ldb + k0 + seg * 8);
        union { uint4 u; __bf16 h[8]; } c; c.u = q;
#pragma unroll
        for (int j = 0; j < 8; ++j) Bs[buf][(seg * 8 + j) * BS + n] = c.h[j];
      }
    } else {                                 // W stored [K,N]: direct copy
#pragma unroll
      for (int it = 0; it < 2; ++it) {
        int id = tid + it * 256;
        int kr = id >> 4, seg = id & 15;
        const uint4 q = *(const uint4*)(Bm + (size_t)(k0 + kr) * g.ldb + nb + seg * 8);
        unsigned int* dst = (unsigned int*)(&Bs[buf][0] + kr * BS + seg * 8);
        dst[0] = q.x; dst[1] = q.y; dst[2] = q.z; dst[3] = q.w;
      }
    }
  };

  CFrag acc[4];
#pragma unroll
  for (int s = 0; s < 4; ++s)
#pragma unroll
    for (int r = 0; r < 8; ++r) acc[s].e[r] = 0.0f;

  stage(0, 0);
  for (int k0 = 0; k0 < g.K; k0 += 32) {
    const int cur = (k0 >> 5) & 1;
    __syncthreads();                       // staged `cur` visible; `cur^1` drained
    if (k0 + 32 < g.K) stage(cur ^ 1, k0 + 32);   // overlap loads with WMMA
    bf16x16 bf = load_b_frag(&Bs[cur][0] + wv * 16, BS);
#pragma unroll
    for (int s = 0; s < 4; ++s) {
      AFrag af = load_a_frag(&As[cur][0] + s * 16 * AS, AS);
      acc[s].v = wmma_bf16(af.v, bf, acc[s].v);
    }
  }

  // ---- epilogues ----
  const int gn = nb + wv * 16 + (lane & 15);
  const int hi = (lane >> 4) & 1;
  float bn;
  if constexpr (MODE <= 1) bn = g.bias[gn]; else bn = bias[gn];

  int which = 0, head = 0, di = 0;
  if constexpr (MODE == 0) {
    which = gn / Ddim;
    int dd = gn - which * Ddim;
    head = dd >> 6; di = dd & 63;
  }

#pragma unroll
  for (int s = 0; s < 4; ++s) {
#pragma unroll
    for (int r = 0; r < 8; ++r) {
      int ml = s * 16 + r + 8 * hi;
      float v = acc[s].e[r];
      if constexpr (MODE == 0) {
        int gm = mb + ml;                       // token index
        float val = v + bn;
        int ss = gm >> 3, bb = gm & 7;          // t = s*B + b
        size_t idx = (((size_t)bb * Hn + head) * Sdim + ss) * HD + di;
        if (which == 0)      g.qo[idx] = f2bf(val * 0.125f);   // hd^-0.5
        else if (which == 1) g.ko[idx] = f2bf(val);
        else                 g.vo[idx] = f2bf(val);
      } else if constexpr (MODE == 1) {
        int gm = mb + ml;
        size_t idx = (size_t)gm * Ddim + gn;
        g.out_f32[idx] = v + bn + g.resid[idx];
      } else if constexpr (MODE == 2) {
        int lm = mb + ml;
        if (lm < cnt_e) {
          float x = v + bn;
          x = x / (1.0f + __expf(-1.702f * x));                // QuickGELU
          g.out_bf16[(size_t)(base + lm) * FFdim + gn] = f2bf(x);
        }
      } else {
        int lm = mb + ml;
        if (lm < cnt_e) {
          int p = base + lm;
          int entry = g.rows[p];
          float cw = g.wts[p];
          g.out_f32[(size_t)entry * Ddim + gn] = cw * (v + bn);
        }
      }
    }
  }
}

// ---------------------------------------------------------------------------
// Flash attention: block = 128 thr (4 waves), 64 q rows / block, one (b,h).
// Double-buffered K/V chunks of 32 keys; online softmax (permlane reductions);
// P transposed through per-wave LDS into A-frag layout.
// ---------------------------------------------------------------------------
__launch_bounds__(128)
__global__ void attn_kernel(const __bf16* __restrict__ qb,
                            const __bf16* __restrict__ kb,
                            const __bf16* __restrict__ vb,
                            __bf16* __restrict__ ob) {
  __shared__ __bf16 Kl[2][32 * 66];
  __shared__ __bf16 Vl[2][32 * 66];
  __shared__ __bf16 Pl[4][16 * 34];

  const int bh = blockIdx.y;                 // 0..95 = b*H + h
  const int bb = bh / Hn, hh = bh % Hn;
  const int tid = threadIdx.x, wv = tid >> 5, lane = tid & 31;
  const int q0 = blockIdx.x * 64 + wv * 16;

  const __bf16* qbase = qb + (size_t)bh * Sdim * HD;
  const __bf16* kbase = kb + (size_t)bh * Sdim * HD;
  const __bf16* vbase = vb + (size_t)bh * Sdim * HD;

  auto stageKV = [&](int buf, int c0) {
#pragma unroll
    for (int it = 0; it < 2; ++it) {
      int id = tid + it * 128;
      int row = id >> 3, seg = id & 7;
      const uint4 qk = ((const uint4*)(kbase + (size_t)(c0 + row) * HD))[seg];
      const uint4 qv = ((const uint4*)(vbase + (size_t)(c0 + row) * HD))[seg];
      unsigned int* dk = (unsigned int*)&Kl[buf][0] + row * 33 + seg * 4;
      unsigned int* dv = (unsigned int*)&Vl[buf][0] + row * 33 + seg * 4;
      dk[0] = qk.x; dk[1] = qk.y; dk[2] = qk.z; dk[3] = qk.w;
      dv[0] = qv.x; dv[1] = qv.y; dv[2] = qv.z; dv[3] = qv.w;
    }
  };

  // load the wave's q fragments (16 rows x 64 dims = 2 A-frags), once
  AFrag qf[2];
  {
    int m = lane & 15, hi = (lane >> 4) & 1;
    const __bf16* p = qbase + (size_t)(q0 + m) * HD;
#pragma unroll
    for (int c = 0; c < 2; ++c) {
      const __bf16* pc = p + c * 32 + hi * 8;
#pragma unroll
      for (int i = 0; i < 8; ++i) { qf[c].e[i] = pc[i]; qf[c].e[i + 8] = pc[i + 16]; }
    }
  }

  CFrag of[4];
#pragma unroll
  for (int t = 0; t < 4; ++t)
#pragma unroll
    for (int r = 0; r < 8; ++r) of[t].e[r] = 0.0f;
  float mrun[8], lrun[8];
#pragma unroll
  for (int r = 0; r < 8; ++r) { mrun[r] = -3e38f; lrun[r] = 0.0f; }

  stageKV(0, 0);
  for (int c0 = 0; c0 < Sdim; c0 += 32) {
    const int cur = (c0 >> 5) & 1;
    __syncthreads();
    if (c0 + 32 < Sdim) stageKV(cur ^ 1, c0 + 32);   // overlap with WMMA/softmax

    // S = q @ K^T for 32 keys -> two 16x16 fp32 tiles
    CFrag s0, s1;
#pragma unroll
    for (int r = 0; r < 8; ++r) { s0.e[r] = 0.0f; s1.e[r] = 0.0f; }
#pragma unroll
    for (int d = 0; d < 2; ++d) {
      bf16x16 kb0 = load_bT_frag(&Kl[cur][0] + d * 32, 66);            // keys 0..15
      bf16x16 kb1 = load_bT_frag(&Kl[cur][0] + 16 * 66 + d * 32, 66);  // keys 16..31
      s0.v = wmma_bf16(qf[d].v, kb0, s0.v);
      s1.v = wmma_bf16(qf[d].v, kb1, s1.v);
    }

    // online softmax (rows live at vgpr r, half-wave hi); permlane reductions
    float al[8];
#pragma unroll
    for (int r = 0; r < 8; ++r) {
      float mc = rmax16(fmaxf(s0.e[r], s1.e[r]));
      float nm = fmaxf(mrun[r], mc);
      al[r] = __expf(mrun[r] - nm);
      s0.e[r] = __expf(s0.e[r] - nm);
      s1.e[r] = __expf(s1.e[r] - nm);
      float rs = rsum16(s0.e[r] + s1.e[r]);
      lrun[r] = lrun[r] * al[r] + rs;
      mrun[r] = nm;
    }
#pragma unroll
    for (int t = 0; t < 4; ++t)
#pragma unroll
      for (int r = 0; r < 8; ++r) of[t].e[r] *= al[r];

    // transpose P through per-wave LDS into A-frag layout
    __bf16* pl = &Pl[wv][0];
    {
      int n = lane & 15, hi = (lane >> 4) & 1;
#pragma unroll
      for (int r = 0; r < 8; ++r) {
        int m = r + 8 * hi;
        pl[m * 34 + n]      = f2bf(s0.e[r]);
        pl[m * 34 + 16 + n] = f2bf(s1.e[r]);
      }
    }
    AFrag pf;
    {
      int m = lane & 15, hi = (lane >> 4) & 1;
      const __bf16* p = pl + m * 34 + hi * 8;
#pragma unroll
      for (int i = 0; i < 8; ++i) { pf.e[i] = p[i]; pf.e[i + 8] = p[i + 16]; }
    }

    // O += P @ V  (four 16-col tiles of head dim)
#pragma unroll
    for (int t = 0; t < 4; ++t) {
      bf16x16 vf = load_b_frag(&Vl[cur][0] + t * 16, 66);
      of[t].v = wmma_bf16(pf.v, vf, of[t].v);
    }
  }

  // normalize and store o[token, h*64 + d] bf16 (token-major for out-proj)
  {
    int n = lane & 15, hi = (lane >> 4) & 1;
#pragma unroll
    for (int t = 0; t < 4; ++t)
#pragma unroll
      for (int r = 0; r < 8; ++r) {
        int m = r + 8 * hi;
        int sq = q0 + m;
        float val = of[t].e[r] / lrun[r];
        ob[(size_t)(sq * Bdim + bb) * Ddim + hh * 64 + t * 16 + n] = f2bf(val);
      }
  }
}

// ---------------------------------------------------------------------------
// Routing: one wave per token. logits = h2 . gate^T, softmax, top-2, weights.
// ---------------------------------------------------------------------------
__launch_bounds__(256)
__global__ void routing_kernel(const __bf16* __restrict__ h2,
                               const float* __restrict__ gate,
                               float* __restrict__ logits_out,
                               int* __restrict__ rt_sel, float* __restrict__ rt_w,
                               int* __restrict__ cnt) {
  int wv = threadIdx.x >> 5, lane = threadIdx.x & 31;
  int t = blockIdx.x * 8 + wv;
  const __bf16* row = h2 + (size_t)t * Ddim;
  float xv[24];
#pragma unroll
  for (int j = 0; j < 24; ++j) xv[j] = bf2f(row[lane + 32 * j]);
  float lg[En];
#pragma unroll
  for (int e = 0; e < En; ++e) {
    const float* g = gate + (size_t)e * Ddim;
    float s = 0.f;
#pragma unroll
    for (int j = 0; j < 24; ++j) s += xv[j] * g[lane + 32 * j];
    lg[e] = rsum32(s);
  }
  if (lane == 0) {
    float mx = lg[0];
#pragma unroll
    for (int e = 1; e < En; ++e) mx = fmaxf(mx, lg[e]);
    float p[En]; float sum = 0.f;
#pragma unroll
    for (int e = 0; e < En; ++e) { p[e] = __expf(lg[e] - mx); sum += p[e]; }
    float inv = 1.0f / sum;
#pragma unroll
    for (int e = 0; e < En; ++e) { p[e] *= inv; logits_out[(size_t)t * En + e] = lg[e]; }
    int e0 = 0;
    for (int e = 1; e < En; ++e) if (p[e] > p[e0]) e0 = e;
    int e1 = (e0 == 0) ? 1 : 0;
    for (int e = 0; e < En; ++e) if (e != e0 && p[e] > p[e1]) e1 = e;
    float w0 = p[e0], w1 = p[e1];
    float rn = 1.0f / (w0 + w1);
    rt_sel[2 * t] = e0; rt_sel[2 * t + 1] = e1;
    rt_w[2 * t] = w0 * rn; rt_w[2 * t + 1] = w1 * rn;
    atomicAdd(&cnt[e0], 1);
    atomicAdd(&cnt[e1], 1);
  }
}

__global__ void route_init_kernel(int* cnt, int* cursor) {
  int t = threadIdx.x;
  if (t < En) { cnt[t] = 0; cursor[t] = 0; }
}

__global__ void route_scan_kernel(const int* cnt, int* offp, int* cursor) {
  if (threadIdx.x == 0) {
    int a = 0;
    for (int e = 0; e < En; ++e) { offp[e] = a; a += cnt[e]; }
  }
  if (threadIdx.x < En) cursor[threadIdx.x] = 0;
}

__global__ void route_build_kernel(const int* __restrict__ rt_sel,
                                   const float* __restrict__ rt_w,
                                   const int* __restrict__ offp, int* cursor,
                                   int* __restrict__ rows, float* __restrict__ wts) {
  int t = blockIdx.x * blockDim.x + threadIdx.x;
  if (t >= Tn) return;
#pragma unroll
  for (int slot = 0; slot < 2; ++slot) {
    int e = rt_sel[2 * t + slot];
    int i = atomicAdd(&cursor[e], 1);
    int p = offp[e] + i;
    rows[p] = 2 * t + slot;
    wts[p] = rt_w[2 * t + slot];
  }
}

// x_out = x1 + slot0 + slot1 (combine weights already applied)
__global__ void combine_kernel(const float* __restrict__ x1,
                               const float* __restrict__ slots,
                               float* __restrict__ xout) {
  long n = (long)Tn * Ddim;
  long i = (long)blockIdx.x * blockDim.x + threadIdx.x;
  long stride = (long)gridDim.x * blockDim.x;
  for (; i < n; i += stride) {
    long t = i / Ddim, d = i - t * Ddim;
    xout[i] = x1[i] + slots[(size_t)(2 * t) * Ddim + d] +
              slots[(size_t)(2 * t + 1) * Ddim + d];
  }
}

// ---------------------------------------------------------------------------
extern "C" void kernel_launch(void* const* d_in, const int* in_sizes, int n_in,
                              void* d_out, int out_size, void* d_ws, size_t ws_size,
                              hipStream_t stream) {
  (void)in_sizes; (void)n_in; (void)out_size; (void)ws_size;
  const float* x          = (const float*)d_in[0];
  const float* ln1_w      = (const float*)d_in[1];
  const float* ln1_b      = (const float*)d_in[2];
  const float* in_proj_w  = (const float*)d_in[3];
  const float* in_proj_b  = (const float*)d_in[4];
  const float* out_proj_w = (const float*)d_in[5];
  const float* out_proj_b = (const float*)d_in[6];
  const float* ln2_w      = (const float*)d_in[7];
  const float* ln2_b      = (const float*)d_in[8];
  const float* gate_w     = (const float*)d_in[9];
  const float* w1         = (const float*)d_in[10];
  const float* b1         = (const float*)d_in[11];
  const float* w2         = (const float*)d_in[12];
  const float* b2         = (const float*)d_in[13];

  char* ws = (char*)d_ws;
  size_t off = 0;
  auto alloc = [&](size_t bytes) -> char* {
    off = (off + 255) & ~size_t(255);
    char* p = ws + off; off += bytes; return p;
  };
  // ~332 MB total workspace
  __bf16* Wqkv = (__bf16*)alloc((size_t)3 * Ddim * Ddim * 2);
  __bf16* Wout = (__bf16*)alloc((size_t)Ddim * Ddim * 2);
  __bf16* W1b  = (__bf16*)alloc((size_t)En * Ddim * FFdim * 2);
  __bf16* W2b  = (__bf16*)alloc((size_t)En * FFdim * Ddim * 2);
  __bf16* hbuf = (__bf16*)alloc((size_t)Tn * Ddim * 2);
  __bf16* qbuf = (__bf16*)alloc((size_t)Bdim * Hn * Sdim * HD * 2);
  __bf16* kbuf = (__bf16*)alloc((size_t)Bdim * Hn * Sdim * HD * 2);
  __bf16* vbuf = (__bf16*)alloc((size_t)Bdim * Hn * Sdim * HD * 2);
  __bf16* obuf = (__bf16*)alloc((size_t)Tn * Ddim * 2);
  float*  x1   = (float*)alloc((size_t)Tn * Ddim * 4);
  __bf16* h2   = (__bf16*)alloc((size_t)Tn * Ddim * 2);
  __bf16* mid  = (__bf16*)alloc((size_t)2 * Tn * FFdim * 2);
  float*  slots= (float*)alloc((size_t)2 * Tn * Ddim * 4);
  int*    cnt  = (int*)alloc(En * 4);
  int*    eoff = (int*)alloc(En * 4);
  int*    curs = (int*)alloc(En * 4);
  int*    rows = (int*)alloc((size_t)2 * Tn * 4);
  float*  wts  = (float*)alloc((size_t)2 * Tn * 4);
  int*    rt_sel = (int*)alloc((size_t)2 * Tn * 4);
  float*  rt_w   = (float*)alloc((size_t)2 * Tn * 4);

  float* xout = (float*)d_out;
  float* logits_out = xout + (size_t)Tn * Ddim;

  // 1) convert weights to bf16 (weights fit in 192MB L2; one pass at HBM speed)
  f32_to_bf16_kernel<<<4096, 256, 0, stream>>>(in_proj_w, Wqkv, (long)3 * Ddim * Ddim);
  f32_to_bf16_kernel<<<2048, 256, 0, stream>>>(out_proj_w, Wout, (long)Ddim * Ddim);
  f32_to_bf16_kernel<<<8192, 256, 0, stream>>>(w1, W1b, (long)En * Ddim * FFdim);
  f32_to_bf16_kernel<<<8192, 256, 0, stream>>>(w2, W2b, (long)En * FFdim * Ddim);
  route_init_kernel<<<1, 32, 0, stream>>>(cnt, curs);

  // 2) LN1
  ln_kernel<<<Tn / 8, 256, 0, stream>>>(x, ln1_w, ln1_b, hbuf);

  // 3) QKV projection
  {
    GArgs g{}; g.A = hbuf; g.B = Wqkv; g.bias = in_proj_b;
    g.qo = qbuf; g.ko = kbuf; g.vo = vbuf;
    g.N = 3 * Ddim; g.K = Ddim; g.lda = Ddim; g.ldb = Ddim;
    gemm_kernel<0><<<dim3((3 * Ddim) / 128, Tn / 64), 256, 0, stream>>>(g);
  }

  // 4) attention
  attn_kernel<<<dim3(Sdim / 64, Bdim * Hn), 128, 0, stream>>>(qbuf, kbuf, vbuf, obuf);

  // 5) output projection + residual -> x1
  {
    GArgs g{}; g.A = obuf; g.B = Wout; g.bias = out_proj_b; g.resid = x;
    g.out_f32 = x1;
    g.N = Ddim; g.K = Ddim; g.lda = Ddim; g.ldb = Ddim;
    gemm_kernel<1><<<dim3(Ddim / 128, Tn / 64), 256, 0, stream>>>(g);
  }

  // 6) LN2 + routing + dispatch lists
  ln_kernel<<<Tn / 8, 256, 0, stream>>>(x1, ln2_w, ln2_b, h2);
  routing_kernel<<<Tn / 8, 256, 0, stream>>>(h2, gate_w, logits_out, rt_sel, rt_w, cnt);
  route_scan_kernel<<<1, 32, 0, stream>>>(cnt, eoff, curs);
  route_build_kernel<<<(Tn + 255) / 256, 256, 0, stream>>>(rt_sel, rt_w, eoff, curs,
                                                           rows, wts);

  // 7) gathered expert FFN (top-2 sparse: ~155 GF instead of 618 GF dense)
  {
    GArgs g{}; g.A = h2; g.B = W1b; g.bias = b1;
    g.out_bf16 = mid; g.rows = rows; g.wts = wts; g.cnt = cnt; g.off = eoff;
    g.N = FFdim; g.K = Ddim; g.lda = Ddim; g.ldb = FFdim;
    gemm_kernel<2><<<dim3(FFdim / 128, Tn / 64, En), 256, 0, stream>>>(g);
  }
  {
    GArgs g{}; g.A = mid; g.B = W2b; g.bias = b2;
    g.out_f32 = slots; g.rows = rows; g.wts = wts; g.cnt = cnt; g.off = eoff;
    g.N = Ddim; g.K = FFdim; g.lda = FFdim; g.ldb = Ddim;
    gemm_kernel<3><<<dim3(Ddim / 128, Tn / 64, En), 256, 0, stream>>>(g);
  }

  // 8) combine -> final output
  combine_kernel<<<4096, 256, 0, stream>>>(x1, slots, xout);
}
